// q_ResBlock_30562987278759
// MI455X (gfx1250) — 8-bit matrix operand order repaired and checked by reading, compiled, not run on hardware
//
#include <hip/hip_runtime.h>
#include <stdint.h>

#define BB 128
#define SS 196
#define DD 384
#define CC 1536
#define SP 256
#define MM (BB * SS)
#define QMAXF 127.0f
#define EPSF 1e-8f

typedef int v8i __attribute__((ext_vector_type(8)));

__device__ __forceinline__ float scale_from(const float* mb, int slot) {
  return fmaxf(mb[slot] * (1.0f / QMAXF), EPSF);
}
__device__ __forceinline__ float qclampf(float v) {
  return fmaxf(fminf(v, QMAXF), -QMAXF);
}

__device__ __forceinline__ void block_max_atomic(float v, float* slot) {
  __shared__ float red[8];
  #pragma unroll
  for (int off = 16; off; off >>= 1)
    v = fmaxf(v, __shfl_xor(v, off, 32));
  int wid = threadIdx.x >> 5;
  int lane = threadIdx.x & 31;
  if (lane == 0) red[wid] = v;
  __syncthreads();
  if (threadIdx.x == 0) {
    int nw = (blockDim.x + 31) >> 5;
    float m = red[0];
    for (int i = 1; i < nw; ++i) m = fmaxf(m, red[i]);
    atomicMax((unsigned int*)slot, __float_as_uint(m));
  }
}

__device__ __forceinline__ v8i a_frag(const signed char* __restrict__ A, int lda,
                                      int m0, int k0, int l, int hi) {
  const signed char* row = A + (size_t)(m0 + l) * lda + (k0 + (hi << 4));
  v8i r;
  r[0] = *(const int*)(row + 0);
  r[1] = *(const int*)(row + 4);
  r[2] = *(const int*)(row + 8);
  r[3] = *(const int*)(row + 12);
  r[4] = *(const int*)(row + 32);
  r[5] = *(const int*)(row + 36);
  r[6] = *(const int*)(row + 40);
  r[7] = *(const int*)(row + 44);
  return r;
}

__device__ __forceinline__ v8i b_frag(const signed char* __restrict__ Bm, int ldb,
                                      int n0, int k0, int l, int hi) {
  const signed char* col = Bm + (size_t)(n0 + l) * ldb + (k0 + (hi << 4));
  const int4 lo = *(const int4*)(col);
  const int4 hv = *(const int4*)(col + 32);
  v8i r;
  r[0] = lo.x; r[1] = lo.y; r[2] = lo.z; r[3] = lo.w;
  r[4] = hv.x; r[5] = hv.y; r[6] = hv.z; r[7] = hv.w;
  return r;
}

__global__ void init_maxbuf_kernel(float* mb) {
  if (threadIdx.x < 8) mb[threadIdx.x] = 0.0f;
}

__global__ void quant_rows_kernel(const float* __restrict__ W, int rows_real,
                                  int cols, int cols_pad,
                                  signed char* __restrict__ Wq,
                                  float* __restrict__ srow) {
  int r = blockIdx.x;
  signed char* out = Wq + (size_t)r * cols_pad;
  if (r >= rows_real) {
    for (int c = threadIdx.x; c < cols_pad; c += blockDim.x) out[c] = 0;
    if (threadIdx.x == 0) srow[r] = 1.0f;
    return;
  }
  const float* w = W + (size_t)r * cols;
  __shared__ float red[8];
  __shared__ float s_sh;
  float m = 0.0f;
  for (int c = threadIdx.x; c < cols; c += blockDim.x) m = fmaxf(m, fabsf(w[c]));
  #pragma unroll
  for (int off = 16; off; off >>= 1) m = fmaxf(m, __shfl_xor(m, off, 32));
  int wid = threadIdx.x >> 5, lane = threadIdx.x & 31;
  if (lane == 0) red[wid] = m;
  __syncthreads();
  if (threadIdx.x == 0) {
    int nw = (blockDim.x + 31) >> 5;
    float mm = red[0];
    for (int i = 1; i < nw; ++i) mm = fmaxf(mm, red[i]);
    float s = fmaxf(mm * (1.0f / QMAXF), EPSF);
    s_sh = s;
    srow[r] = s;
  }
  __syncthreads();
  float s = s_sh;
  for (int c = threadIdx.x; c < cols_pad; c += blockDim.x) {
    float q = 0.0f;
    if (c < cols) q = qclampf(rintf(w[c] / s));
    out[c] = (signed char)(int)q;
  }
}

__global__ void act1_max_kernel(const float* __restrict__ x,
                                const float* __restrict__ a1,
                                const float* __restrict__ b1, float* mb) {
  const size_t n = (size_t)MM * DD;
  float lm = 0.0f;
  for (size_t i = (size_t)blockIdx.x * blockDim.x + threadIdx.x; i < n;
       i += (size_t)gridDim.x * blockDim.x) {
    int d = (int)(i % DD);
    lm = fmaxf(lm, fabsf(x[i] * a1[d] + b1[d]));
  }
  block_max_atomic(lm, mb + 0);
}

__global__ void act1_quant_kernel(const float* __restrict__ x,
                                  const float* __restrict__ a1,
                                  const float* __restrict__ b1,
                                  const float* __restrict__ mb,
                                  signed char* __restrict__ h1q) {
  const size_t n = (size_t)BB * DD * SP;
  float sa = scale_from(mb, 0);
  for (size_t i = (size_t)blockIdx.x * blockDim.x + threadIdx.x; i < n;
       i += (size_t)gridDim.x * blockDim.x) {
    int b = (int)(i / ((size_t)DD * SP));
    int rem = (int)(i % ((size_t)DD * SP));
    int d = rem / SP;
    int s = rem % SP;
    signed char q = 0;
    if (s < SS) {
      float h = x[((size_t)b * SS + s) * DD + d] * a1[d] + b1[d];
      q = (signed char)(int)qclampf(rintf(h / sa));
    }
    h1q[i] = q;
  }
}

__global__ void act2_kernel(const float* __restrict__ h2,
                            const float* __restrict__ a2,
                            const float* __restrict__ b2, float* mb,
                            float* __restrict__ h2q) {
  const size_t n = (size_t)MM * DD;
  float s2 = scale_from(mb, 1);
  float lm = 0.0f;
  for (size_t i = (size_t)blockIdx.x * blockDim.x + threadIdx.x; i < n;
       i += (size_t)gridDim.x * blockDim.x) {
    float q = qclampf(rintf(h2[i] / s2)) * s2;
    h2q[i] = q;
    int d = (int)(i % DD);
    lm = fmaxf(lm, fabsf(q * a2[d] + b2[d]));
  }
  block_max_atomic(lm, mb + 2);
}

__global__ void act3_quant_kernel(const float* __restrict__ h2q,
                                  const float* __restrict__ a2,
                                  const float* __restrict__ b2,
                                  const float* __restrict__ mb,
                                  signed char* __restrict__ gq) {
  const size_t n = (size_t)MM * DD;
  float sg = scale_from(mb, 2);
  for (size_t i = (size_t)blockIdx.x * blockDim.x + threadIdx.x; i < n;
       i += (size_t)gridDim.x * blockDim.x) {
    int d = (int)(i % DD);
    float g = h2q[i] * a2[d] + b2[d];
    gq[i] = (signed char)(int)qclampf(rintf(g / sg));
  }
}

__global__ void act4_quant_kernel(const float* __restrict__ gel,
                                  const float* __restrict__ mb,
                                  signed char* __restrict__ gelq) {
  const size_t n = (size_t)MM * CC;
  float s4 = scale_from(mb, 3);
  for (size_t i = (size_t)blockIdx.x * blockDim.x + threadIdx.x; i < n;
       i += (size_t)gridDim.x * blockDim.x) {
    gelq[i] = (signed char)(int)qclampf(rintf(gel[i] / s4));
  }
}

__global__ void final_quant_kernel(const float* __restrict__ h3,
                                   const float* __restrict__ mb,
                                   float* __restrict__ out, unsigned long long n) {
  float s5 = scale_from(mb, 4);
  for (size_t i = (size_t)blockIdx.x * blockDim.x + threadIdx.x; i < n;
       i += (size_t)gridDim.x * blockDim.x) {
    out[i] = qclampf(rintf(h3[i] / s5)) * s5;
  }
  if (blockIdx.x == 0 && threadIdx.x == 0) out[n] = s5;
}

__global__ void __launch_bounds__(128) gemm_tok_kernel(
    const signed char* __restrict__ Aq,
    const signed char* __restrict__ Bq,
    const float* __restrict__ s_wt, float* mb,
    const float* __restrict__ x, const float* __restrict__ btok,
    const float* __restrict__ ls1, float* __restrict__ h2) {
  int lane = threadIdx.x & 31;
  int l = lane & 15, hi = lane >> 4;
  int wave = threadIdx.x >> 5;
  int wm = wave >> 1, wn = wave & 1;
  int m0 = blockIdx.y * 64 + wm * 32;
  int n0 = blockIdx.x * 128 + wn * 64;
  int b = blockIdx.z;
  const signed char* Bp = Bq + (size_t)b * DD * SP;

  v8i acc[2][4];
  #pragma unroll
  for (int i = 0; i < 2; ++i)
    #pragma unroll
    for (int j = 0; j < 4; ++j) acc[i][j] = (v8i){0, 0, 0, 0, 0, 0, 0, 0};

  #pragma unroll
  for (int k0 = 0; k0 < SP; k0 += 64) {
    v8i a0 = a_frag(Aq, SP, m0, k0, l, hi);
    v8i a1 = a_frag(Aq, SP, m0 + 16, k0, l, hi);
    #pragma unroll
    for (int j = 0; j < 4; ++j) {
      v8i bj = b_frag(Bp, SP, n0 + j * 16, k0, l, hi);
      acc[0][j] = __builtin_amdgcn_wmma_i32_16x16x64_iu8(true, a0, true, bj,
                                                         acc[0][j], false, false);
      acc[1][j] = __builtin_amdgcn_wmma_i32_16x16x64_iu8(true, a1, true, bj,
                                                         acc[1][j], false, false);
    }
  }

  float sa = scale_from(mb, 0);
  float lm = 0.0f;
  #pragma unroll
  for (int i = 0; i < 2; ++i)
    #pragma unroll
    for (int j = 0; j < 4; ++j)
      #pragma unroll
      for (int r = 0; r < 8; ++r) {
        int t = m0 + i * 16 + hi * 8 + r;
        int d = n0 + j * 16 + l;
        if (t < SS) {
          size_t idx = ((size_t)b * SS + t) * DD + d;
          float y = (float)acc[i][j][r] * (sa * s_wt[t]) + btok[t];
          float v = x[idx] + ls1[d] * y;
          h2[idx] = v;
          lm = fmaxf(lm, fabsf(v));
        }
      }
  block_max_atomic(lm, mb + 1);
}

__global__ void __launch_bounds__(128) gemm_fc1_kernel(
    const signed char* __restrict__ Aq,
    const signed char* __restrict__ Bq,
    const float* __restrict__ s_w1, float* mb,
    const float* __restrict__ bfc1, float* __restrict__ gel) {
  int lane = threadIdx.x & 31;
  int l = lane & 15, hi = lane >> 4;
  int wave = threadIdx.x >> 5;
  int wm = wave >> 1, wn = wave & 1;
  int m0 = blockIdx.y * 64 + wm * 32;
  int n0 = blockIdx.x * 128 + wn * 64;

  v8i acc[2][4];
  #pragma unroll
  for (int i = 0; i < 2; ++i)
    #pragma unroll
    for (int j = 0; j < 4; ++j) acc[i][j] = (v8i){0, 0, 0, 0, 0, 0, 0, 0};

  #pragma unroll
  for (int k0 = 0; k0 < DD; k0 += 64) {
    v8i a0 = a_frag(Aq, DD, m0, k0, l, hi);
    v8i a1 = a_frag(Aq, DD, m0 + 16, k0, l, hi);
    #pragma unroll
    for (int j = 0; j < 4; ++j) {
      v8i bj = b_frag(Bq, DD, n0 + j * 16, k0, l, hi);
      acc[0][j] = __builtin_amdgcn_wmma_i32_16x16x64_iu8(true, a0, true, bj,
                                                         acc[0][j], false, false);
      acc[1][j] = __builtin_amdgcn_wmma_i32_16x16x64_iu8(true, a1, true, bj,
                                                         acc[1][j], false, false);
    }
  }

  float sg = scale_from(mb, 2);
  float lm = 0.0f;
  const float kAlpha = 0.7978845608028654f;
  #pragma unroll
  for (int i = 0; i < 2; ++i)
    #pragma unroll
    for (int j = 0; j < 4; ++j)
      #pragma unroll
      for (int r = 0; r < 8; ++r) {
        int m = m0 + i * 16 + hi * 8 + r;
        int c = n0 + j * 16 + l;
        float v = (float)acc[i][j][r] * (sg * s_w1[c]) + bfc1[c];
        float g = 0.5f * v * (1.0f + tanhf(kAlpha * (v + 0.044715f * v * v * v)));
        gel[(size_t)m * CC + c] = g;
        lm = fmaxf(lm, fabsf(g));
      }
  block_max_atomic(lm, mb + 3);
}

__global__ void __launch_bounds__(128) gemm_fc2_kernel(
    const signed char* __restrict__ Aq,
    const signed char* __restrict__ Bq,
    const float* __restrict__ s_w2, float* mb,
    const float* __restrict__ bfc2, const float* __restrict__ ls2,
    const float* __restrict__ h2q, float* __restrict__ h3) {
  int lane = threadIdx.x & 31;
  int l = lane & 15, hi = lane >> 4;
  int wave = threadIdx.x >> 5;
  int wm = wave >> 1, wn = wave & 1;
  int m0 = blockIdx.y * 64 + wm * 32;
  int n0 = blockIdx.x * 128 + wn * 64;

  v8i acc[2][4];
  #pragma unroll
  for (int i = 0; i < 2; ++i)
    #pragma unroll
    for (int j = 0; j < 4; ++j) acc[i][j] = (v8i){0, 0, 0, 0, 0, 0, 0, 0};

  for (int k0 = 0; k0 < CC; k0 += 64) {
    v8i a0 = a_frag(Aq, CC, m0, k0, l, hi);
    v8i a1 = a_frag(Aq, CC, m0 + 16, k0, l, hi);
    #pragma unroll
    for (int j = 0; j < 4; ++j) {
      v8i bj = b_frag(Bq, CC, n0 + j * 16, k0, l, hi);
      acc[0][j] = __builtin_amdgcn_wmma_i32_16x16x64_iu8(true, a0, true, bj,
                                                         acc[0][j], false, false);
      acc[1][j] = __builtin_amdgcn_wmma_i32_16x16x64_iu8(true, a1, true, bj,
                                                         acc[1][j], false, false);
    }
  }

  float s4 = scale_from(mb, 3);
  float lm = 0.0f;
  #pragma unroll
  for (int i = 0; i < 2; ++i)
    #pragma unroll
    for (int j = 0; j < 4; ++j)
      #pragma unroll
      for (int r = 0; r < 8; ++r) {
        int m = m0 + i * 16 + hi * 8 + r;
        int n = n0 + j * 16 + l;
        size_t idx = (size_t)m * DD + n;
        float v = (float)acc[i][j][r] * (s4 * s_w2[n]) + bfc2[n];
        float o = h2q[idx] + ls2[n] * v;
        h3[idx] = o;
        lm = fmaxf(lm, fabsf(o));
      }
  block_max_atomic(lm, mb + 4);
}

static inline unsigned nblocks(size_t n) {
  size_t b = (n + 255) / 256;
  return (unsigned)(b < 32768 ? b : 32768);
}

extern "C" void kernel_launch(void* const* d_in, const int* in_sizes, int n_in,
                              void* d_out, int out_size, void* d_ws, size_t ws_size,
                              hipStream_t stream) {
  (void)in_sizes; (void)n_in; (void)out_size; (void)ws_size;
  const float* x    = (const float*)d_in[0];
  const float* a1   = (const float*)d_in[2];
  const float* b1   = (const float*)d_in[3];
  const float* a2   = (const float*)d_in[4];
  const float* b2   = (const float*)d_in[5];
  const float* wt   = (const float*)d_in[8];
  const float* btok = (const float*)d_in[9];
  const float* w1   = (const float*)d_in[10];
  const float* bfc1 = (const float*)d_in[11];
  const float* w2   = (const float*)d_in[12];
  const float* bfc2 = (const float*)d_in[13];
  const float* ls1  = (const float*)d_in[14];
  const float* ls2  = (const float*)d_in[15];
  float* out = (float*)d_out;

  char* base = (char*)d_ws;
  size_t off = 0;
  auto take = [&](size_t bytes) -> void* {
    void* p = base + off;
    off += (bytes + 255) & ~(size_t)255;
    return p;
  };
  float* mb        = (float*)take(8 * sizeof(float));
  float* s_wt      = (float*)take(256 * sizeof(float));
  float* s_w1      = (float*)take((size_t)CC * sizeof(float));
  float* s_w2      = (float*)take((size_t)DD * sizeof(float));
  signed char* wqt = (signed char*)take((size_t)256 * SP);
  signed char* wq1 = (signed char*)take((size_t)CC * DD);
  signed char* wq2 = (signed char*)take((size_t)DD * CC);
  signed char* h1q = (signed char*)take((size_t)BB * DD * SP);
  float* h2        = (float*)take((size_t)MM * DD * sizeof(float));
  float* h2q       = (float*)take((size_t)MM * DD * sizeof(float));
  signed char* gq  = (signed char*)take((size_t)MM * DD);
  float* gel       = (float*)take((size_t)MM * CC * sizeof(float));
  signed char* glq = (signed char*)take((size_t)MM * CC);
  float* h3 = h2;

  const size_t nBSD = (size_t)MM * DD;

  init_maxbuf_kernel<<<1, 32, 0, stream>>>(mb);
  quant_rows_kernel<<<256, 256, 0, stream>>>(wt, SS, SS, SP, wqt, s_wt);
  quant_rows_kernel<<<CC, 256, 0, stream>>>(w1, CC, DD, DD, wq1, s_w1);
  quant_rows_kernel<<<DD, 256, 0, stream>>>(w2, DD, CC, CC, wq2, s_w2);

  act1_max_kernel<<<nblocks(nBSD), 256, 0, stream>>>(x, a1, b1, mb);
  act1_quant_kernel<<<nblocks((size_t)BB * DD * SP), 256, 0, stream>>>(x, a1, b1, mb, h1q);

  gemm_tok_kernel<<<dim3(DD / 128, 4, BB), 128, 0, stream>>>(wqt, h1q, s_wt, mb,
                                                             x, btok, ls1, h2);

  act2_kernel<<<nblocks(nBSD), 256, 0, stream>>>(h2, a2, b2, mb, h2q);
  act3_quant_kernel<<<nblocks(nBSD), 256, 0, stream>>>(h2q, a2, b2, mb, gq);

  gemm_fc1_kernel<<<dim3(CC / 128, MM / 64), 128, 0, stream>>>(gq, wq1, s_w1, mb,
                                                               bfc1, gel);
  act4_quant_kernel<<<nblocks((size_t)MM * CC), 256, 0, stream>>>(gel, mb, glq);

  gemm_fc2_kernel<<<dim3(DD / 128, MM / 64), 128, 0, stream>>>(glq, wq2, s_w2, mb,
                                                               bfc2, ls2, h2q, h3);

  final_quant_kernel<<<nblocks(nBSD), 256, 0, stream>>>(h3, mb, out,
                                                        (unsigned long long)nBSD);
}
